// CrossScaleCoupling_17136919511708
// MI455X (gfx1250) — compile-verified, matrix operand order checked by reading, not run on hardware
//
#include <hip/hip_runtime.h>
#include <hip/hip_bf16.h>

// ---------------------------------------------------------------------------
// CrossScaleCoupling for MI455X (gfx1250), wave32 + WMMA bf16 + async LDS.
// ---------------------------------------------------------------------------

typedef __bf16 bf16;
typedef bf16  v16bf __attribute__((ext_vector_type(16)));
typedef bf16  v8bf  __attribute__((ext_vector_type(8)));
typedef bf16  v4bf  __attribute__((ext_vector_type(4)));
typedef bf16  v2bf  __attribute__((ext_vector_type(2)));
typedef float v8f   __attribute__((ext_vector_type(8)));
typedef int   v4i_  __attribute__((ext_vector_type(4)));

constexpr int kB  = 4;
constexpr int kTU = 2048;
constexpr int kTM = 256;
constexpr int kD  = 1024;
constexpr int kH  = 16;

static __device__ __forceinline__ v8f z8() {
  v8f z = {0.f, 0.f, 0.f, 0.f, 0.f, 0.f, 0.f, 0.f};
  return z;
}

// NT fragment load from global (A and B operands of C = X * Y^T):
// lane l<16  -> row row0+l,    cols k0 .. k0+15
// lane l>=16 -> row row0+l-16, cols k0+16 .. k0+31
static __device__ __forceinline__ v16bf ld_frag(const bf16* __restrict__ base,
                                                int ld, int row0, int k0) {
  const int l = threadIdx.x & 31;
  const bf16* p = base + (long)(row0 + (l & 15)) * ld + (k0 + ((l >> 4) << 4));
  return *(const v16bf*)p;
}

// Same fragment pattern out of a staged LDS tile (row stride 32 elements).
static __device__ __forceinline__ v16bf ld_frag_lds(const bf16* s, int row0) {
  const int l = threadIdx.x & 31;
  return *(const v16bf*)(s + (row0 + (l & 15)) * 32 + ((l >> 4) << 4));
}

static __device__ __forceinline__ v8f wmma_bf16(v16bf a, v16bf b, v8f c) {
  return __builtin_amdgcn_wmma_f32_16x16x32_bf16(false, a, false, b,
                                                 (short)0, c, false, false);
}

static __device__ __forceinline__ float sigmoidf_(float x) {
  return 1.f / (1.f + __expf(-x));
}

// ---------------------------------------------------------------------------
// Async global -> LDS staging (16 bytes per call).  The gfx1250 builtin takes
// (global v4i* src, local v4i* dst, imm offset, imm cpol).  Guarded so the
// file also compiles on toolchains without the builtin.
// ---------------------------------------------------------------------------
static __device__ __forceinline__ void stage16(const bf16* __restrict__ g, bf16* l) {
#if defined(__HIP_DEVICE_COMPILE__) && __has_builtin(__builtin_amdgcn_global_load_async_to_lds_b128)
  __builtin_amdgcn_global_load_async_to_lds_b128(
      (__attribute__((address_space(1))) v4i_*)g,
      (__attribute__((address_space(3))) v4i_*)l, 0, 0);
#else
  *(v8bf*)l = *(const v8bf*)g;
#endif
}

static __device__ __forceinline__ void wait_async4() {
#if defined(__HIP_DEVICE_COMPILE__) && __has_builtin(__builtin_amdgcn_s_wait_asynccnt)
  __builtin_amdgcn_s_wait_asynccnt(4);
#elif defined(__HIP_DEVICE_COMPILE__)
  asm volatile("s_wait_asynccnt 0x4" ::: "memory");
#endif
}
static __device__ __forceinline__ void wait_async0() {
#if defined(__HIP_DEVICE_COMPILE__) && __has_builtin(__builtin_amdgcn_s_wait_asynccnt)
  __builtin_amdgcn_s_wait_asynccnt(0);
#elif defined(__HIP_DEVICE_COMPILE__)
  asm volatile("s_wait_asynccnt 0x0" ::: "memory");
#endif
}

// ---------------------------------------------------------------------------
// Staged NT GEMM core: one 64x64 tile per 128-thread block (4 waves, 2x2 of
// 32x32).  Double-buffered LDS staging of X (64xk32) and W (64xk32) tiles via
// async LDS loads; WMMA fragments consumed from LDS.  K = kD = 1024.
// Fills acc[2][2]; wave sub-tile origin returned in (wi, wj).
// ---------------------------------------------------------------------------
static __device__ __forceinline__ void gemm_nt_64x64(
    const bf16* __restrict__ X, const bf16* __restrict__ W,
    int r0, int c0, v8f acc[2][2], int* wi_out, int* wj_out) {
  __shared__ __align__(64) bf16 Xs[2][64 * 32];
  __shared__ __align__(64) bf16 Ws[2][64 * 32];
  const int tid  = threadIdx.x;
  const int wave = tid >> 5;
  const int wi = (wave >> 1) * 32;  // row half within 64-tile
  const int wj = (wave & 1) * 32;   // col half within 64-tile
  *wi_out = wi; *wj_out = wj;

  // Stage one 64x32 k-slab of X and W: 256 16B chunks each, 2 per thread.
  auto stage = [&](int k, int buf) {
    for (int i = 0; i < 2; ++i) {
      int c = tid + 128 * i;
      int row = c >> 2, seg = (c & 3) * 8;
      stage16(X + (long)(r0 + row) * kD + k + seg, &Xs[buf][row * 32 + seg]);
      stage16(W + (long)(c0 + row) * kD + k + seg, &Ws[buf][row * 32 + seg]);
    }
  };

  acc[0][0] = z8(); acc[0][1] = z8(); acc[1][0] = z8(); acc[1][1] = z8();
  stage(0, 0);
  for (int k = 0, buf = 0; k < kD; k += 32, buf ^= 1) {
    if (k + 32 < kD) { stage(k + 32, buf ^ 1); wait_async4(); }
    else             { wait_async0(); }
    __syncthreads();  // stage k visible to all 4 waves
    v16bf x0 = ld_frag_lds(&Xs[buf][0], wi);
    v16bf x1 = ld_frag_lds(&Xs[buf][0], wi + 16);
    v16bf w0 = ld_frag_lds(&Ws[buf][0], wj);
    v16bf w1 = ld_frag_lds(&Ws[buf][0], wj + 16);
    acc[0][0] = wmma_bf16(x0, w0, acc[0][0]);
    acc[0][1] = wmma_bf16(x0, w1, acc[0][1]);
    acc[1][0] = wmma_bf16(x1, w0, acc[1][0]);
    acc[1][1] = wmma_bf16(x1, w1, acc[1][1]);
    __syncthreads();  // everyone done with buf before it is restaged
  }
}

// ---------------------------------------------------------------------------
// fp32 -> bf16 conversion (weights)
// ---------------------------------------------------------------------------
__global__ void k_cvt(const float* __restrict__ src, bf16* __restrict__ dst, int n) {
  int i = blockIdx.x * blockDim.x + threadIdx.x;
  if (i < n) dst[i] = (bf16)src[i];
}

// ---------------------------------------------------------------------------
// Fused balance_norm + gate + phase vector.  One block per token, 256 thr.
// ---------------------------------------------------------------------------
__global__ void __launch_bounds__(256) k_norm(
    const float* __restrict__ x,
    const float* __restrict__ w, const float* __restrict__ bvec,
    const float* __restrict__ balw,
    const float* __restrict__ gw, const float* __restrict__ gb,
    bf16* __restrict__ nrm, bf16* __restrict__ ph, float* __restrict__ gout) {
  __shared__ float sh[24];
  const int t = blockIdx.x;
  const int tid = threadIdx.x;
  const float4 v = ((const float4*)(x + (long)t * kD))[tid];
  float s  = v.x + v.y + v.z + v.w;
  float sq = v.x * v.x + v.y * v.y + v.z * v.z + v.w * v.w;
  float sqf = (tid < 128) ? sq : 0.f;
  float r0 = s, r1 = sq, r2 = sqf;
  for (int m = 16; m; m >>= 1) {
    r0 += __shfl_xor(r0, m, 32);
    r1 += __shfl_xor(r1, m, 32);
    r2 += __shfl_xor(r2, m, 32);
  }
  if ((tid & 31) == 0) {
    int wid = tid >> 5;
    sh[wid] = r0; sh[8 + wid] = r1; sh[16 + wid] = r2;
  }
  __syncthreads();
  float sum = 0.f, sumsq = 0.f, sqfirst = 0.f;
  for (int i = 0; i < 8; ++i) { sum += sh[i]; sumsq += sh[8 + i]; sqfirst += sh[16 + i]; }
  const float mu   = sum * (1.f / 1024.f);
  const float var  = sumsq * (1.f / 1024.f) - mu * mu;
  const float rstd = rsqrtf(var + 1e-6f);
  const float e1   = sqfirst * (1.f / 512.f);
  const float e2   = (sumsq - sqfirst) * (1.f / 512.f);
  const float corr = ((tid < 128) ? -1.f : 1.f) * (e1 - e2) * balw[0];
  const int i0 = tid * 4;
  float n0 = (v.x - mu) * rstd * w[i0 + 0] + bvec[i0 + 0] + corr;
  float n1 = (v.y - mu) * rstd * w[i0 + 1] + bvec[i0 + 1] + corr;
  float n2 = (v.z - mu) * rstd * w[i0 + 2] + bvec[i0 + 2] + corr;
  float n3 = (v.w - mu) * rstd * w[i0 + 3] + bvec[i0 + 3] + corr;
  v4bf nv = {(bf16)n0, (bf16)n1, (bf16)n2, (bf16)n3};
  *(v4bf*)(nrm + (long)t * kD + i0) = nv;
  // cos(atan2(y+eps,x+eps)) = (x+eps)/r, sin = (y+eps)/r -- no trig needed
  float a0 = n0 + 1e-8f, b0 = n1 + 1e-8f;
  float a1 = n2 + 1e-8f, b1 = n3 + 1e-8f;
  float ir0 = rsqrtf(a0 * a0 + b0 * b0);
  float ir1 = rsqrtf(a1 * a1 + b1 * b1);
  v2bf cv = {(bf16)(a0 * ir0), (bf16)(a1 * ir1)};
  v2bf sv = {(bf16)(b0 * ir0), (bf16)(b1 * ir1)};
  *(v2bf*)(ph + (long)t * kD + 2 * tid) = cv;
  *(v2bf*)(ph + (long)t * kD + 512 + 2 * tid) = sv;
  float gp = n0 * gw[i0] + n1 * gw[i0 + 1] + n2 * gw[i0 + 2] + n3 * gw[i0 + 3];
  __syncthreads();
  for (int m = 16; m; m >>= 1) gp += __shfl_xor(gp, m, 32);
  if ((tid & 31) == 0) sh[tid >> 5] = gp;
  __syncthreads();
  if (tid == 0) {
    float g = 0.f;
    for (int i = 0; i < 8; ++i) g += sh[i];
    gout[t] = sigmoidf_(g + gb[0]);
  }
}

// ---------------------------------------------------------------------------
// Phase resonance: R = Pm * Pu^T / 512 per batch via the staged GEMM core,
// fused EMA epilogue writes r_conv_new and r_emer_new (= transpose).
// ---------------------------------------------------------------------------
__global__ void __launch_bounds__(128) k_reson(
    const bf16* __restrict__ phm, const bf16* __restrict__ phu,
    const float* __restrict__ accC, const float* __restrict__ accE,
    float* __restrict__ outC, float* __restrict__ outE) {
  const int tm_t = kTM / 64, tu_t = kTU / 64;  // 4, 32
  const int b   = blockIdx.x / (tm_t * tu_t);
  const int rem = blockIdx.x % (tm_t * tu_t);
  const int m0 = (rem / tu_t) * 64, u0 = (rem % tu_t) * 64;
  const bf16* A  = phm + (long)b * kTM * kD;
  const bf16* Bm = phu + (long)b * kTU * kD;
  v8f acc[2][2];
  int wi, wj;
  gemm_nt_64x64(A, Bm, m0, u0, acc, &wi, &wj);
  const int l = threadIdx.x & 31, rbv = (l >> 4) << 3, cl = l & 15;
  const float inv = 1.f / 512.f;
  for (int i = 0; i < 2; ++i)
    for (int j = 0; j < 2; ++j)
      for (int r = 0; r < 8; ++r) {
        int m = m0 + wi + 16 * i + rbv + r;
        int u = u0 + wj + 16 * j + cl;
        float rinst = acc[i][j][r] * inv;
        long ic = ((long)b * kTM + m) * kTU + u;
        long ie = ((long)b * kTU + u) * kTM + m;
        outC[ic] = 0.7f * accC[ic] + 0.3f * rinst;
        outE[ie] = 0.7f * accE[ie] + 0.3f * rinst;
      }
}

// ---------------------------------------------------------------------------
// Row-mean -> sigmoid(alpha*mean) scalars (deterministic block reduction).
// ---------------------------------------------------------------------------
__global__ void __launch_bounds__(256) k_rowscal(
    const float* __restrict__ mat, int cols,
    const float* __restrict__ alpha, float* __restrict__ out) {
  __shared__ float sh[8];
  const int row = blockIdx.x;
  const float* r = mat + (long)row * cols;
  float s = 0.f;
  for (int i = threadIdx.x; i < cols; i += 256) s += r[i];
  for (int m = 16; m; m >>= 1) s += __shfl_xor(s, m, 32);
  if ((threadIdx.x & 31) == 0) sh[threadIdx.x >> 5] = s;
  __syncthreads();
  if (threadIdx.x == 0) {
    float t = 0.f;
    for (int i = 0; i < 8; ++i) t += sh[i];
    out[row] = sigmoidf_(alpha[0] * t / (float)cols);
  }
}

// ---------------------------------------------------------------------------
// Projection GEMM: C = X (rows x 1024) * W^T, staged core, bf16 epilogue.
// mode 0: row-major out.  mode 1: per-head transposed out [b][h][d][t].
// ---------------------------------------------------------------------------
__global__ void __launch_bounds__(128) k_gemm_nt(
    const bf16* __restrict__ X, const bf16* __restrict__ W,
    int rows, int mode, int Tseq, bf16* __restrict__ out) {
  const int tiles_n = kD / 64;  // 16
  const int r0 = (blockIdx.x / tiles_n) * 64, c0 = (blockIdx.x % tiles_n) * 64;
  (void)rows;
  v8f acc[2][2];
  int wi, wj;
  gemm_nt_64x64(X, W, r0, c0, acc, &wi, &wj);
  const int l = threadIdx.x & 31, rbv = (l >> 4) << 3, cl = l & 15;
  for (int i = 0; i < 2; ++i)
    for (int j = 0; j < 2; ++j)
      for (int r = 0; r < 8; ++r) {
        int row = r0 + wi + 16 * i + rbv + r;
        int col = c0 + wj + 16 * j + cl;
        bf16 vv = (bf16)acc[i][j][r];
        if (mode == 0) {
          out[(long)row * kD + col] = vv;
        } else {
          int bb = row / Tseq, t = row % Tseq;
          int h = col >> 6, d = col & 63;
          out[(((long)bb * kH + h) * 64 + d) * Tseq + t] = vv;
        }
      }
}

// ---------------------------------------------------------------------------
// Output projection + residual + resonance scalar (fp32 epilogue to d_out).
// ---------------------------------------------------------------------------
__global__ void __launch_bounds__(128) k_outproj(
    const bf16* __restrict__ X, const bf16* __restrict__ W, int rows,
    const float* __restrict__ orig, const float* __restrict__ scal,
    float* __restrict__ out) {
  const int tiles_n = kD / 64;
  const int r0 = (blockIdx.x / tiles_n) * 64, c0 = (blockIdx.x % tiles_n) * 64;
  (void)rows;
  v8f acc[2][2];
  int wi, wj;
  gemm_nt_64x64(X, W, r0, c0, acc, &wi, &wj);
  const int l = threadIdx.x & 31, rbv = (l >> 4) << 3, cl = l & 15;
  for (int i = 0; i < 2; ++i)
    for (int j = 0; j < 2; ++j)
      for (int r = 0; r < 8; ++r) {
        int row = r0 + wi + 16 * i + rbv + r;
        int col = c0 + wj + 16 * j + cl;
        long idx = (long)row * kD + col;
        out[idx] = orig[idx] + acc[i][j][r] * scal[row];
      }
}

// ---------------------------------------------------------------------------
// Attention pass 1: per-row max, Z = sum exp, Z' = sum exp*eg (online).
// Stores m[row] and folded constant C = ag / (ag*Z' + 1e-8*Z).
// One wave = 16 query rows, streaming K in 32-token chunks.
// ---------------------------------------------------------------------------
__global__ void __launch_bounds__(128) k_attn_stats(
    const bf16* __restrict__ Q, const bf16* __restrict__ K,
    int Tq, int Tk,
    const float* __restrict__ gq, const float* __restrict__ gk,
    float* __restrict__ stM, float* __restrict__ stC) {
  const int gid = blockIdx.x * 4 + (threadIdx.x >> 5);
  const int qtiles = Tq / 16;
  const int bh = gid / qtiles, qt = gid % qtiles;
  const int b = bh >> 4, h = bh & 15;
  const int q0 = qt * 16;
  const int l = threadIdx.x & 31, cl = l & 15;
  const bf16* Qb = Q + (long)b * Tq * kD + h * 64;
  const bf16* Kb = K + (long)b * Tk * kD + h * 64;
  const v16bf qa0 = ld_frag(Qb, kD, q0, 0);
  const v16bf qa1 = ld_frag(Qb, kD, q0, 32);
  float runm[8], runz[8], runzp[8];
  for (int r = 0; r < 8; ++r) { runm[r] = -1e30f; runz[r] = 0.f; runzp[r] = 0.f; }
  const float sc = 0.125f;  // 1/sqrt(64)
  for (int kt = 0; kt < Tk; kt += 32) {
    v16bf b00 = ld_frag(Kb, kD, kt, 0);
    v16bf b01 = ld_frag(Kb, kD, kt, 32);
    v16bf b10 = ld_frag(Kb, kD, kt + 16, 0);
    v16bf b11 = ld_frag(Kb, kD, kt + 16, 32);
    v8f s0 = z8(), s1 = z8();
    s0 = wmma_bf16(qa0, b00, s0); s0 = wmma_bf16(qa1, b01, s0);
    s1 = wmma_bf16(qa0, b10, s1); s1 = wmma_bf16(qa1, b11, s1);
    const float eg0 = gk[(long)b * Tk + kt + cl];
    const float eg1 = gk[(long)b * Tk + kt + 16 + cl];
    for (int r = 0; r < 8; ++r) {
      float v0 = s0[r] * sc, v1 = s1[r] * sc;
      float mx = fmaxf(v0, v1);
      for (int m = 8; m; m >>= 1) mx = fmaxf(mx, __shfl_xor(mx, m, 32));
      float nm = fmaxf(runm[r], mx);
      float f  = __expf(runm[r] - nm);
      float e0 = __expf(v0 - nm), e1 = __expf(v1 - nm);
      float pz = e0 + e1, pzp = e0 * eg0 + e1 * eg1;
      for (int m = 8; m; m >>= 1) {
        pz  += __shfl_xor(pz, m, 32);
        pzp += __shfl_xor(pzp, m, 32);
      }
      runz[r]  = runz[r]  * f + pz;
      runzp[r] = runzp[r] * f + pzp;
      runm[r]  = nm;
    }
  }
  if (cl == 0) {
    const int rbv = (l >> 4) << 3;
    for (int r = 0; r < 8; ++r) {
      int q = q0 + rbv + r;
      long idx = (long)bh * Tq + q;
      float ag = gq[(long)b * Tq + q];
      stM[idx] = runm[r];
      stC[idx] = ag / (ag * runzp[r] + 1e-8f * runz[r]);
    }
  }
}

// ---------------------------------------------------------------------------
// Attention pass 2: recompute scores, P = exp(s-m)*eg*C, C-layout -> A-layout
// via LDS (explicit s_wait_dscnt), then ctx = P * V via NT WMMA on V^T.
// ---------------------------------------------------------------------------
__global__ void __launch_bounds__(128) k_attn_ctx(
    const bf16* __restrict__ Q, const bf16* __restrict__ K,
    const bf16* __restrict__ Vt, int Tq, int Tk,
    const float* __restrict__ gk,
    const float* __restrict__ stM, const float* __restrict__ stC,
    bf16* __restrict__ ctx) {
  __shared__ __align__(32) bf16 pl[4][16 * 32];
  const int wave = threadIdx.x >> 5;
  const int gid = blockIdx.x * 4 + wave;
  const int qtiles = Tq / 16;
  const int bh = gid / qtiles, qt = gid % qtiles;
  const int b = bh >> 4, h = bh & 15;
  const int q0 = qt * 16;
  const int l = threadIdx.x & 31, cl = l & 15, rbv = (l >> 4) << 3;
  const bf16* Qb = Q + (long)b * Tq * kD + h * 64;
  const bf16* Kb = K + (long)b * Tk * kD + h * 64;
  const bf16* Vb = Vt + (long)bh * 64 * Tk;
  const v16bf qa0 = ld_frag(Qb, kD, q0, 0);
  const v16bf qa1 = ld_frag(Qb, kD, q0, 32);
  float mrow[8], crow[8];
  for (int r = 0; r < 8; ++r) {
    long idx = (long)bh * Tq + q0 + rbv + r;
    mrow[r] = stM[idx];
    crow[r] = stC[idx];
  }
  v8f o0 = z8(), o1 = z8(), o2 = z8(), o3 = z8();
  bf16* myp = &pl[wave][0];
  const float sc = 0.125f;
  for (int kt = 0; kt < Tk; kt += 32) {
    v16bf b00 = ld_frag(Kb, kD, kt, 0);
    v16bf b01 = ld_frag(Kb, kD, kt, 32);
    v16bf b10 = ld_frag(Kb, kD, kt + 16, 0);
    v16bf b11 = ld_frag(Kb, kD, kt + 16, 32);
    v8f s0 = z8(), s1 = z8();
    s0 = wmma_bf16(qa0, b00, s0); s0 = wmma_bf16(qa1, b01, s0);
    s1 = wmma_bf16(qa0, b10, s1); s1 = wmma_bf16(qa1, b11, s1);
    const float eg0 = gk[(long)b * Tk + kt + cl];
    const float eg1 = gk[(long)b * Tk + kt + 16 + cl];
    for (int r = 0; r < 8; ++r) {
      float p0 = __expf(s0[r] * sc - mrow[r]) * eg0 * crow[r];
      float p1 = __expf(s1[r] * sc - mrow[r]) * eg1 * crow[r];
      myp[(rbv + r) * 32 + cl]      = (bf16)p0;
      myp[(rbv + r) * 32 + 16 + cl] = (bf16)p1;
    }
#if defined(__HIP_DEVICE_COMPILE__)
    asm volatile("s_wait_dscnt 0x0" ::: "memory");
#endif
    v16bf pa = *(const v16bf*)(myp + cl * 32 + ((l >> 4) << 4));
    o0 = wmma_bf16(pa, ld_frag(Vb, Tk, 0,  kt), o0);
    o1 = wmma_bf16(pa, ld_frag(Vb, Tk, 16, kt), o1);
    o2 = wmma_bf16(pa, ld_frag(Vb, Tk, 32, kt), o2);
    o3 = wmma_bf16(pa, ld_frag(Vb, Tk, 48, kt), o3);
  }
  v8f oo[4] = {o0, o1, o2, o3};
  for (int g = 0; g < 4; ++g)
    for (int r = 0; r < 8; ++r)
      ctx[((long)b * Tq + q0 + rbv + r) * kD + h * 64 + g * 16 + cl] = (bf16)oo[g][r];
}

// ---------------------------------------------------------------------------
// Host launcher
// ---------------------------------------------------------------------------
extern "C" void kernel_launch(void* const* d_in, const int* in_sizes, int n_in,
                              void* d_out, int out_size, void* d_ws, size_t ws_size,
                              hipStream_t stream) {
  (void)in_sizes; (void)n_in; (void)out_size; (void)ws_size;
  const float* o_micro    = (const float*)d_in[0];
  const float* o_macro    = (const float*)d_in[1];
  const float* r_conv_acc = (const float*)d_in[2];
  const float* r_emer_acc = (const float*)d_in[3];
  const float* ln_u_w = (const float*)d_in[4];
  const float* ln_u_b = (const float*)d_in[5];
  const float* bw_u   = (const float*)d_in[6];
  const float* ln_m_w = (const float*)d_in[7];
  const float* ln_m_b = (const float*)d_in[8];
  const float* bw_m   = (const float*)d_in[9];
  const float* gu_w = (const float*)d_in[10];
  const float* gu_b = (const float*)d_in[11];
  const float* gm_w = (const float*)d_in[12];
  const float* gm_b = (const float*)d_in[13];
  const float* res_alpha = (const float*)d_in[22];
  float* out = (float*)d_out;

  const int NU = kB * kTU;  // 8192 micro tokens
  const int NM = kB * kTM;  // 1024 macro tokens
  const int WE = kD * kD;   // 1M elements per weight matrix

  char* wsb = (char*)d_ws;
  size_t off = 0;
  auto take = [&](size_t bytes) -> char* {
    char* p = wsb + off;
    off = (off + bytes + 255) & ~(size_t)255;
    return p;
  };
  bf16* nm_u = (bf16*)take((size_t)NU * kD * 2);
  bf16* nm_m = (bf16*)take((size_t)NM * kD * 2);
  bf16* ph_u = (bf16*)take((size_t)NU * kD * 2);
  bf16* ph_m = (bf16*)take((size_t)NM * kD * 2);
  bf16* wb   = (bf16*)take((size_t)8 * WE * 2);
  float* g_u = (float*)take((size_t)NU * 4);
  float* g_m = (float*)take((size_t)NM * 4);
  bf16* q_m  = (bf16*)take((size_t)NM * kD * 2);
  bf16* k_u  = (bf16*)take((size_t)NU * kD * 2);
  bf16* vt_u = (bf16*)take((size_t)NU * kD * 2);
  bf16* q_u  = (bf16*)take((size_t)NU * kD * 2);
  bf16* k_m  = (bf16*)take((size_t)NM * kD * 2);
  bf16* vt_m = (bf16*)take((size_t)NM * kD * 2);
  float* stM_c = (float*)take((size_t)kB * kH * kTM * 4);
  float* stC_c = (float*)take((size_t)kB * kH * kTM * 4);
  float* stM_e = (float*)take((size_t)kB * kH * kTU * 4);
  float* stC_e = (float*)take((size_t)kB * kH * kTU * 4);
  bf16* ctx_m  = (bf16*)take((size_t)NM * kD * 2);
  bf16* ctx_u  = (bf16*)take((size_t)NU * kD * 2);
  float* scal_c = (float*)take((size_t)NM * 4);
  float* scal_e = (float*)take((size_t)NU * 4);

  float* out_micro = out;
  float* out_macro = out_micro + (size_t)NU * kD;
  float* out_rconv = out_macro + (size_t)NM * kD;
  float* out_remer = out_rconv + (size_t)kB * kTM * kTU;

  // 1. convert the 8 weight matrices to bf16
  for (int i = 0; i < 8; ++i)
    k_cvt<<<WE / 256, 256, 0, stream>>>((const float*)d_in[14 + i], wb + (size_t)i * WE, WE);
  bf16* w_cq = wb + 0 * (size_t)WE;
  bf16* w_ck = wb + 1 * (size_t)WE;
  bf16* w_cv = wb + 2 * (size_t)WE;
  bf16* w_co = wb + 3 * (size_t)WE;
  bf16* w_eq = wb + 4 * (size_t)WE;
  bf16* w_ek = wb + 5 * (size_t)WE;
  bf16* w_ev = wb + 6 * (size_t)WE;
  bf16* w_eo = wb + 7 * (size_t)WE;

  // 2. fused balance_norm + gate + phase
  k_norm<<<NU, 256, 0, stream>>>(o_micro, ln_u_w, ln_u_b, bw_u, gu_w, gu_b, nm_u, ph_u, g_u);
  k_norm<<<NM, 256, 0, stream>>>(o_macro, ln_m_w, ln_m_b, bw_m, gm_w, gm_b, nm_m, ph_m, g_m);

  // 3. phase resonance GEMM + EMA (writes r_conv_new and r_emer_new)
  k_reson<<<kB * (kTM / 64) * (kTU / 64), 128, 0, stream>>>(
      ph_m, ph_u, r_conv_acc, r_emer_acc, out_rconv, out_remer);

  // 4. resonance scalars
  k_rowscal<<<NM, 256, 0, stream>>>(out_rconv, kTU, res_alpha, scal_c);
  k_rowscal<<<NU, 256, 0, stream>>>(out_remer, kTM, res_alpha, scal_e);

  // 5. QKV projections (V stored per-head transposed)
  auto ggrid = [](int rows) { return (rows / 64) * (kD / 64); };
  k_gemm_nt<<<ggrid(NM), 128, 0, stream>>>(nm_m, w_cq, NM, 0, kTM, q_m);
  k_gemm_nt<<<ggrid(NU), 128, 0, stream>>>(nm_u, w_ck, NU, 0, kTU, k_u);
  k_gemm_nt<<<ggrid(NU), 128, 0, stream>>>(nm_u, w_cv, NU, 1, kTU, vt_u);
  k_gemm_nt<<<ggrid(NU), 128, 0, stream>>>(nm_u, w_eq, NU, 0, kTU, q_u);
  k_gemm_nt<<<ggrid(NM), 128, 0, stream>>>(nm_m, w_ek, NM, 0, kTM, k_m);
  k_gemm_nt<<<ggrid(NM), 128, 0, stream>>>(nm_m, w_ev, NM, 1, kTM, vt_m);

  // 6. attention (conv: macro queries over micro keys; emer: transposed roles)
  const int wconv = kB * kH * (kTM / 16);  // 1024 waves
  const int wemer = kB * kH * (kTU / 16);  // 8192 waves
  k_attn_stats<<<wconv / 4, 128, 0, stream>>>(q_m, k_u, kTM, kTU, g_m, g_u, stM_c, stC_c);
  k_attn_ctx  <<<wconv / 4, 128, 0, stream>>>(q_m, k_u, vt_u, kTM, kTU, g_u, stM_c, stC_c, ctx_m);
  k_attn_stats<<<wemer / 4, 128, 0, stream>>>(q_u, k_m, kTU, kTM, g_u, g_m, stM_e, stC_e);
  k_attn_ctx  <<<wemer / 4, 128, 0, stream>>>(q_u, k_m, vt_m, kTU, kTM, g_m, stM_e, stC_e, ctx_u);

  // 7. output projections + residual + resonance scalar
  k_outproj<<<ggrid(NM), 128, 0, stream>>>(ctx_m, w_co, NM, o_macro, scal_c, out_macro);
  k_outproj<<<ggrid(NU), 128, 0, stream>>>(ctx_u, w_eo, NU, o_micro, scal_e, out_micro);
}
